// BuildingGenRegModel_7533372637727
// MI455X (gfx1250) — matrix operand order repaired and checked by reading, not compiled, not run on hardware
//
#include <hip/hip_runtime.h>
#include <math.h>

// GraphSAGE (max aggr) forward for MI455X / gfx1250.
// Memory-bound on 4 segment-max passes (~16 GB @ 23.3 TB/s). Dense math runs
// through v_wmma_f32_16x16x32_f16 (f32 accumulate) with:
//   - weights pre-packed to f16 [Wl|Wr] once per launch (no cvt in hot loop)
//   - A tiles converted f32->f16 once per block into padded LDS, fragments
//     read back as 16B ds loads (no redundant per-wave conversion)

#define N_NODES 50000
#define CH      128
#define KCAT    256                    // agg(128) || self(128)
#define NF      (N_NODES * CH)         // 6,400,000 floats per feature buffer
#define WELEMS  (CH * KCAT)            // 32768 f16 per packed layer weight
#define LDS_PAD 8
#define LDS_STRIDE (KCAT + LDS_PAD)    // 264 halves = 528 B row stride

typedef _Float16 v16h __attribute__((ext_vector_type(16)));
typedef _Float16 v8h  __attribute__((ext_vector_type(8)));
typedef float    v8f  __attribute__((ext_vector_type(8)));

// f32 atomic max via sign-split int/uint atomics (valid for any mix of signs,
// including -inf initialization).
__device__ __forceinline__ void atomic_max_f32(float* addr, float val) {
  if (val >= 0.0f) atomicMax((int*)addr, __float_as_int(val));
  else             atomicMin((unsigned int*)addr, __float_as_uint(val));
}

__global__ void __launch_bounds__(256) fill_neg_inf(float* __restrict__ p, int n) {
  int i = blockIdx.x * blockDim.x + threadIdx.x;
  if (i < n) ((unsigned int*)p)[i] = 0xff800000u;  // -inf
}

// One wave (32 lanes) per edge: lane loads float4 of the 128-ch source row and
// max-scatters it into agg[dst] with sign-split int/uint atomics.
__global__ void __launch_bounds__(256) seg_max_kernel(
    const float* __restrict__ x, const int* __restrict__ src,
    const int* __restrict__ dst, float* __restrict__ agg, int n_edges) {
  int e = (int)((blockIdx.x * blockDim.x + threadIdx.x) >> 5);
  if (e >= n_edges) return;
  int lane = threadIdx.x & 31;
  int s = __builtin_amdgcn_readfirstlane(src[e]);
  int d = __builtin_amdgcn_readfirstlane(dst[e]);
  float4 v = ((const float4*)(x + (long)s * CH))[lane];
  float* a = agg + (long)d * CH + lane * 4;
  atomic_max_f32(a + 0, v.x);
  atomic_max_f32(a + 1, v.y);
  atomic_max_f32(a + 2, v.z);
  atomic_max_f32(a + 3, v.w);
}

// Pack one layer's weights into f16 [o][k] with k = [Wl row | Wr row], K=256.
__global__ void __launch_bounds__(256) pack_weights(
    const float* __restrict__ Wl, const float* __restrict__ Wr,
    _Float16* __restrict__ out) {
  int i = blockIdx.x * blockDim.x + threadIdx.x;   // 0..32767
  if (i >= WELEMS) return;
  int o = i >> 8, k = i & 255;
  float v = (k < CH) ? Wl[o * CH + k] : Wr[o * CH + (k - CH)];
  out[i] = (_Float16)v;
}

// B fragment from packed f16 weights (B = Wcat^T, 32x16 per step).
// lane n = lane&15, kg = lane>>4: halves e hold K = kbase + kg*16 + e,
// i.e. 32 contiguous bytes from row (j0+n) of Wcat. No conversions.
__device__ __forceinline__ v16h load_b_frag(const _Float16* __restrict__ Wcat,
                                            int j0, int kbase, int lane) {
  const _Float16* p = Wcat + (long)(j0 + (lane & 15)) * KCAT + kbase +
                      ((lane >> 4) * 16);
  return *(const v16h*)p;
}

// A fragment from LDS (16x32 f16, contiguous k order, same as B):
// lane group g: VGPR0-3 hold K = kbase+g*16..+8, VGPR4-7 hold K = kbase+g*16+8..+8.
__device__ __forceinline__ v16h load_a_frag(const _Float16* __restrict__ As,
                                            int mr, int kbase, int g) {
  const _Float16* p = As + mr * LDS_STRIDE + kbase + g * 16;
  v8h lo = *(const v8h*)(p);        // 16B ds load
  v8h hi = *(const v8h*)(p + 8);    // 16B ds load
  v16h a;
#pragma unroll
  for (int e = 0; e < 8; ++e) { a[e] = lo[e]; a[e + 8] = hi[e]; }
  return a;
}

// Fused SAGE layer: out = relu([agg|self] @ [Wl|Wr].T + b) as one K=256 WMMA
// pass. HEAD=true folds the 128->1 linear head (shfl reduce + atomicAdd into a
// per-node scalar) so the last feature maps never hit HBM.
template <bool HEAD>
__global__ void __launch_bounds__(256) sage_layer_kernel(
    const float* __restrict__ aggIn, const float* __restrict__ selfIn,
    const _Float16* __restrict__ Wcat, const float* __restrict__ bias,
    float* __restrict__ out,          // HEAD ? [N] scalars : [N,128] features
    const float* __restrict__ w3) {   // head weights [128] (HEAD only)
  __shared__ _Float16 As[16 * LDS_STRIDE];        // 16 rows x 256 K, padded

  // Cooperative f32->f16 tile stage: thread t converts 16 contiguous elements
  // of row t/16 (cols t%16*16). Folds where(isfinite, agg, 0): -inf -> 0.
  const int m0   = blockIdx.x * 16;               // 50000 = 16*3125, exact
  {
    const int t    = threadIdx.x;
    const int row  = t >> 4;
    const int colb = (t & 15) * 16;
    const float* srcp = (colb < CH)
        ? (aggIn  + (long)(m0 + row) * CH + colb)
        : (selfIn + (long)(m0 + row) * CH + (colb - CH));
    _Float16* dstp = As + row * LDS_STRIDE + colb;
#pragma unroll
    for (int e = 0; e < 16; ++e) {
      float v = srcp[e];
      dstp[e] = (_Float16)((v > -1.0e30f) ? v : 0.0f);
    }
  }
  __syncthreads();

  const int lane = threadIdx.x & 31;
  const int wv   = threadIdx.x >> 5;              // 8 waves = 8 column tiles
  const int j0   = wv * 16;
  const int g    = lane >> 4;
  const int mr   = lane & 15;

  v8f c = {};
#pragma unroll
  for (int kk = 0; kk < 8; ++kk) {                // K = 256 in 8 WMMA steps
    v16h a = load_a_frag(As, mr, kk * 32, g);
    v16h b = load_b_frag(Wcat, j0, kk * 32, lane);
    c = __builtin_amdgcn_wmma_f32_16x16x32_f16(false, a, false, b,
                                               (short)0, c, false, false);
  }

  const int   col = j0 + mr;
  const float bj  = bias[col];

  if (!HEAD) {
    // C/D layout: VGPR r -> M = r (lanes 0-15) or r+8 (lanes 16-31), N = lane&15.
#pragma unroll
    for (int r = 0; r < 8; ++r) {
      int m = m0 + r + (g ? 8 : 0);
      out[(long)m * CH + col] = fmaxf(c[r] + bj, 0.0f);
    }
  } else {
    const float wc = w3[col];
#pragma unroll
    for (int r = 0; r < 8; ++r) {
      float p = fmaxf(c[r] + bj, 0.0f) * wc;      // relu then head dot partial
      p += __shfl_xor(p, 1);
      p += __shfl_xor(p, 2);
      p += __shfl_xor(p, 4);
      p += __shfl_xor(p, 8);                      // reduce 16 cols of the tile
      if (mr == 0) atomicAdd(&out[m0 + r + (g ? 8 : 0)], p);
    }
  }
}

// d_out is poisoned by the harness; seed it with the head biases since the
// head-fused layers accumulate with atomicAdd.
__global__ void __launch_bounds__(256) init_out_kernel(
    float* __restrict__ out, const float* __restrict__ rtb,
    const float* __restrict__ mvb, int n) {
  int i = blockIdx.x * blockDim.x + threadIdx.x;
  if (i < n)          out[i] = rtb[0];
  else if (i < 2 * n) out[i] = mvb[0];
}

extern "C" void kernel_launch(void* const* d_in, const int* in_sizes, int n_in,
                              void* d_out, int out_size, void* d_ws, size_t ws_size,
                              hipStream_t stream) {
  const float* x         = (const float*)d_in[0];
  const int*   edge      = (const int*)d_in[1];   // [2, E] int32
  const float* shared_Wl = (const float*)d_in[2];
  const float* shared_b  = (const float*)d_in[3];
  const float* shared_Wr = (const float*)d_in[4];
  const float* rt1_Wl    = (const float*)d_in[5];
  const float* rt1_b     = (const float*)d_in[6];
  const float* rt1_Wr    = (const float*)d_in[7];
  const float* rt2_Wl    = (const float*)d_in[8];
  const float* rt2_b     = (const float*)d_in[9];
  const float* rt2_Wr    = (const float*)d_in[10];
  const float* rt3_W     = (const float*)d_in[11];
  const float* rt3_b     = (const float*)d_in[12];
  const float* mv1_Wl    = (const float*)d_in[13];
  const float* mv1_b     = (const float*)d_in[14];
  const float* mv1_Wr    = (const float*)d_in[15];
  const float* mv2_Wl    = (const float*)d_in[16];
  const float* mv2_b     = (const float*)d_in[17];
  const float* mv2_Wr    = (const float*)d_in[18];
  const float* mv3_W     = (const float*)d_in[19];
  const float* mv3_b     = (const float*)d_in[20];

  const int n_edges = in_sizes[1] / 2;            // 800000
  const int* src = edge;
  const int* dst = edge + n_edges;

  // Workspace: 4 f32 feature buffers (102.4 MB) + 5 packed f16 weights (320 KB)
  float* AGG = (float*)d_ws;
  float* MV  = AGG + NF;
  float* RT  = MV  + NF;
  float* MD  = RT  + NF;
  _Float16* W16 = (_Float16*)(MD + NF);
  _Float16* Wsh  = W16;
  _Float16* Wrt1 = W16 + 1 * WELEMS;
  _Float16* Wmv1 = W16 + 2 * WELEMS;
  _Float16* Wrt2 = W16 + 3 * WELEMS;
  _Float16* Wmv2 = W16 + 4 * WELEMS;
  float* outv = (float*)d_out;                    // [rtAngle | moveDis]

  const int fillBlocks = (NF + 255) / 256;        // 25000
  const int segBlocks  = (n_edges + 7) / 8;       // 8 edges (waves) per block
  const int gemmBlocks = N_NODES / 16;            // 3125 (exact)
  const int packBlocks = (WELEMS + 255) / 256;    // 128

  // Pack all layer weights to f16 [Wl|Wr] (runs concurrently-ish with fill).
  pack_weights<<<packBlocks, 256, 0, stream>>>(shared_Wl, shared_Wr, Wsh);
  pack_weights<<<packBlocks, 256, 0, stream>>>(rt1_Wl, rt1_Wr, Wrt1);
  pack_weights<<<packBlocks, 256, 0, stream>>>(mv1_Wl, mv1_Wr, Wmv1);
  pack_weights<<<packBlocks, 256, 0, stream>>>(rt2_Wl, rt2_Wr, Wrt2);
  pack_weights<<<packBlocks, 256, 0, stream>>>(mv2_Wl, mv2_Wr, Wmv2);

  // shared layer: mv = relu(sage(x))
  fill_neg_inf<<<fillBlocks, 256, 0, stream>>>(AGG, NF);
  seg_max_kernel<<<segBlocks, 256, 0, stream>>>(x, src, dst, AGG, n_edges);
  sage_layer_kernel<false><<<gemmBlocks, 256, 0, stream>>>(
      AGG, x, Wsh, shared_b, MV, nullptr);

  // segment_max(mv) is identical for rt1 and mv1 -> compute once, use twice.
  fill_neg_inf<<<fillBlocks, 256, 0, stream>>>(AGG, NF);
  seg_max_kernel<<<segBlocks, 256, 0, stream>>>(MV, src, dst, AGG, n_edges);
  sage_layer_kernel<false><<<gemmBlocks, 256, 0, stream>>>(
      AGG, MV, Wrt1, rt1_b, RT, nullptr);
  sage_layer_kernel<false><<<gemmBlocks, 256, 0, stream>>>(
      AGG, MV, Wmv1, mv1_b, MD, nullptr);

  // seed both heads with their biases (head layers accumulate via atomicAdd)
  init_out_kernel<<<(2 * N_NODES + 255) / 256, 256, 0, stream>>>(
      outv, rt3_b, mv3_b, N_NODES);

  // rt branch layer 2 + fused head -> rtAngle
  fill_neg_inf<<<fillBlocks, 256, 0, stream>>>(AGG, NF);
  seg_max_kernel<<<segBlocks, 256, 0, stream>>>(RT, src, dst, AGG, n_edges);
  sage_layer_kernel<true><<<gemmBlocks, 256, 0, stream>>>(
      AGG, RT, Wrt2, rt2_b, outv, rt3_W);

  // mv branch layer 2 + fused head -> moveDis
  fill_neg_inf<<<fillBlocks, 256, 0, stream>>>(AGG, NF);
  seg_max_kernel<<<segBlocks, 256, 0, stream>>>(MD, src, dst, AGG, n_edges);
  sage_layer_kernel<true><<<gemmBlocks, 256, 0, stream>>>(
      AGG, MD, Wmv2, mv2_b, outv + N_NODES, mv3_W);
}
